// Autoencoder_64785286693226
// MI455X (gfx1250) — compile-verified, matrix operand order checked by reading, not run on hardware
//
#include <hip/hip_runtime.h>
#include <math.h>

// ---------------------------------------------------------------------------
// Problem constants
// ---------------------------------------------------------------------------
#define NBATCH   256            // n*m = 128*2
#define T_LEN    300
#define V_J      25
#define C_IN     3
#define FLATK    9728           // 64*38*4
#define INV_SQRT_1PEPS 0.9999950000374997f

typedef __attribute__((ext_vector_type(16))) _Float16 v16h;
typedef __attribute__((ext_vector_type(8)))  float    v8f;
typedef __attribute__((ext_vector_type(4)))  float    v4f;

__device__ __forceinline__ float lrelu(float x) { return x >= 0.f ? x : 0.01f * x; }

// ---------------------------------------------------------------------------
// 1) data-BN + layout: x (128,3,300,25,2) -> xm (256,3,300,25)
//    channel = (m*25 + v)*3 + c
// ---------------------------------------------------------------------------
__global__ void prep_xm(const float* __restrict__ x, const float* __restrict__ g,
                        const float* __restrict__ b, float* __restrict__ xm)
{
    int idx = blockIdx.x * blockDim.x + threadIdx.x;
    const int total = NBATCH * C_IN * T_LEN * V_J;
    if (idx >= total) return;
    int v = idx % V_J;   int t2 = idx / V_J;
    int t = t2 % T_LEN;  t2 /= T_LEN;
    int c = t2 % C_IN;   int nm = t2 / C_IN;
    int n = nm >> 1, m = nm & 1;
    int ch = (m * V_J + v) * C_IN + c;
    float val = x[((((size_t)n * C_IN + c) * T_LEN + t) * V_J + v) * 2 + m];
    xm[idx] = val * (g[ch] * INV_SQRT_1PEPS) + b[ch];
}

// dm (256,3,299,25) = xm[:, :, 1:, :] - xm[:, :, :-1, :]
__global__ void prep_dm(const float* __restrict__ xm, float* __restrict__ dm)
{
    int idx = blockIdx.x * blockDim.x + threadIdx.x;
    const int total = NBATCH * C_IN * (T_LEN - 1) * V_J;
    if (idx >= total) return;
    int v = idx % V_J;        int t2 = idx / V_J;
    int t = t2 % (T_LEN - 1); t2 /= (T_LEN - 1);
    int c = t2 % C_IN;        int nm = t2 / C_IN;
    size_t base = ((size_t)(nm * C_IN + c) * T_LEN + t) * V_J + v;
    dm[idx] = xm[base + V_J] - xm[base];
}

// z (256,300): z[:,t<299] = all(dm[:,:,t,:]==0), z[:,299]=0
__global__ void zmask_kernel(const float* __restrict__ dm, int* __restrict__ z)
{
    int idx = blockIdx.x * blockDim.x + threadIdx.x;
    if (idx >= NBATCH * T_LEN) return;
    int t = idx % T_LEN, nm = idx / T_LEN;
    if (t >= T_LEN - 1) { z[idx] = 0; return; }
    int all0 = 1;
    for (int c = 0; c < C_IN && all0; ++c)
        for (int v = 0; v < V_J; ++v)
            if (dm[((size_t)(nm * C_IN + c) * (T_LEN - 1) + t) * V_J + v] != 0.f) { all0 = 0; break; }
    z[idx] = all0;
}

// ---------------------------------------------------------------------------
// Direct conv 3x3 stride 2 pad 1, fused BN + leaky relu.  w: (Cout,Cin,3,3)
// ---------------------------------------------------------------------------
__global__ void conv_s2_bn_lrelu(const float* __restrict__ in, const float* __restrict__ w,
                                 const float* __restrict__ cb, const float* __restrict__ bng,
                                 const float* __restrict__ bnb, float* __restrict__ out,
                                 int NBk, int Cin, int Hin, int Win, int Cout, int Hout, int Wout)
{
    int idx = blockIdx.x * blockDim.x + threadIdx.x;
    int total = NBk * Cout * Hout * Wout;
    if (idx >= total) return;
    int ow = idx % Wout; int t = idx / Wout;
    int oh = t % Hout;   t /= Hout;
    int oc = t % Cout;   int nb = t / Cout;
    float acc = 0.f;
    for (int ic = 0; ic < Cin; ++ic) {
        const float* ip = in + (size_t)(nb * Cin + ic) * Hin * Win;
        const float* wp = w + (size_t)(oc * Cin + ic) * 9;
        #pragma unroll
        for (int kh = 0; kh < 3; ++kh) {
            int ih = oh * 2 - 1 + kh;
            if (ih < 0 || ih >= Hin) continue;
            #pragma unroll
            for (int kw = 0; kw < 3; ++kw) {
                int iw = ow * 2 - 1 + kw;
                if (iw < 0 || iw >= Win) continue;
                acc = fmaf(ip[ih * Win + iw], wp[kh * 3 + kw], acc);
            }
        }
    }
    float s = bng[oc] * INV_SQRT_1PEPS;
    out[idx] = lrelu((acc + cb[oc]) * s + bnb[oc]);
}

// ---------------------------------------------------------------------------
// Transposed conv 3x3 stride 2 pad 1 (gather form), fused BN + leaky relu.
// w layout: (Cin, Cout, 3, 3).   ih = (oh+1-kh)/2 when even.
// ---------------------------------------------------------------------------
__global__ void convT_bn_lrelu(const float* __restrict__ in, const float* __restrict__ w,
                               const float* __restrict__ cb, const float* __restrict__ bng,
                               const float* __restrict__ bnb, float* __restrict__ out,
                               int NBk, int Cin, int Hin, int Win, int Cout, int Hout, int Wout)
{
    int idx = blockIdx.x * blockDim.x + threadIdx.x;
    int total = NBk * Cout * Hout * Wout;
    if (idx >= total) return;
    int ow = idx % Wout; int t = idx / Wout;
    int oh = t % Hout;   t /= Hout;
    int oc = t % Cout;   int nb = t / Cout;
    float acc = 0.f;
    for (int ic = 0; ic < Cin; ++ic) {
        const float* ip = in + (size_t)(nb * Cin + ic) * Hin * Win;
        const float* wp = w + (size_t)(ic * Cout + oc) * 9;
        #pragma unroll
        for (int kh = 0; kh < 3; ++kh) {
            int nh = oh + 1 - kh;
            if (nh & 1) continue;
            int ih = nh >> 1;
            if (ih < 0 || ih >= Hin) continue;
            #pragma unroll
            for (int kw = 0; kw < 3; ++kw) {
                int nw = ow + 1 - kw;
                if (nw & 1) continue;
                int iw = nw >> 1;
                if (iw < 0 || iw >= Win) continue;
                acc = fmaf(ip[ih * Win + iw], wp[kh * 3 + kw], acc);
            }
        }
    }
    float s = bng[oc] * INV_SQRT_1PEPS;
    out[idx] = lrelu((acc + cb[oc]) * s + bnb[oc]);
}

// Final transposed conv (16->3), tanh, cropped directly to (256,3,300,25)
__global__ void convT3_tanh(const float* __restrict__ in, const float* __restrict__ w,
                            const float* __restrict__ cb, float* __restrict__ out)
{
    int idx = blockIdx.x * blockDim.x + threadIdx.x;
    const int total = NBATCH * C_IN * T_LEN * V_J;
    if (idx >= total) return;
    int ow = idx % V_J;  int t = idx / V_J;
    int oh = t % T_LEN;  t /= T_LEN;
    int oc = t % C_IN;   int nb = t / C_IN;
    const int Hin = 152, Win = 16, Cin = 16;
    float acc = 0.f;
    for (int ic = 0; ic < Cin; ++ic) {
        const float* ip = in + (size_t)(nb * Cin + ic) * Hin * Win;
        const float* wp = w + (size_t)(ic * C_IN + oc) * 9;
        #pragma unroll
        for (int kh = 0; kh < 3; ++kh) {
            int nh = oh + 1 - kh;
            if (nh & 1) continue;
            int ih = nh >> 1;
            if (ih < 0 || ih >= Hin) continue;
            #pragma unroll
            for (int kw = 0; kw < 3; ++kw) {
                int nw = ow + 1 - kw;
                if (nw & 1) continue;
                int iw = nw >> 1;
                if (iw < 0 || iw >= Win) continue;
                acc = fmaf(ip[ih * Win + iw], wp[kh * 3 + kw], acc);
            }
        }
    }
    out[idx] = tanhf(acc + cb[oc]);
}

// ---------------------------------------------------------------------------
// WMMA FC layer: out(M,N) = lrelu(A(M,K) @ W(N,K)^T + bias)
// One wave -> 16 rows x 32 cols (two 16x16 D tiles sharing the A fragment);
// 4 waves/block -> 128 output columns.  2 v_wmma per K=32 step.
// A/B fragments per CDNA5 16-bit layout:
//   lanes 0-15 : row/col = lane,    elems 0-7 -> K+0..7,  elems 8-15 -> K+16..23
//   lanes 16-31: row/col = lane-16, elems 0-7 -> K+8..15, elems 8-15 -> K+24..31
// No divergence: EXEC stays all-1s around v_wmma (ISA requirement).
// ---------------------------------------------------------------------------
__global__ void __launch_bounds__(128)
fc_wmma_lrelu(const float* __restrict__ A, const float* __restrict__ W,
              const float* __restrict__ bias, float* __restrict__ out,
              int N, int K)
{
    const int lane  = threadIdx.x & 31;
    const int wave  = threadIdx.x >> 5;
    const int nBase = blockIdx.x * 128 + wave * 32;   // 2 N-tiles per wave
    const int mBase = blockIdx.y * 16;
    const int hl    = lane >> 4;       // 0 or 1
    const int id16  = lane & 15;
    const int kOff  = hl * 8;

    const float* arow  = A + (size_t)(mBase + id16) * K + kOff;
    const float* brow0 = W + (size_t)(nBase + id16) * K + kOff;
    const float* brow1 = W + (size_t)(nBase + 16 + id16) * K + kOff;

    v8f acc0 = {}, acc1 = {};
    for (int k0 = 0; k0 < K; k0 += 32) {
        if (k0 + 256 < K) {   // stream-ahead into GL2/L0 (global_prefetch_b8)
            __builtin_prefetch(arow  + k0 + 256, 0, 1);
            __builtin_prefetch(brow0 + k0 + 256, 0, 1);
            __builtin_prefetch(brow1 + k0 + 256, 0, 1);
        }
        v4f a0 = *(const v4f*)(arow + k0);
        v4f a1 = *(const v4f*)(arow + k0 + 4);
        v4f a2 = *(const v4f*)(arow + k0 + 16);
        v4f a3 = *(const v4f*)(arow + k0 + 20);
        v4f p0 = *(const v4f*)(brow0 + k0);
        v4f p1 = *(const v4f*)(brow0 + k0 + 4);
        v4f p2 = *(const v4f*)(brow0 + k0 + 16);
        v4f p3 = *(const v4f*)(brow0 + k0 + 20);
        v4f q0 = *(const v4f*)(brow1 + k0);
        v4f q1 = *(const v4f*)(brow1 + k0 + 4);
        v4f q2 = *(const v4f*)(brow1 + k0 + 16);
        v4f q3 = *(const v4f*)(brow1 + k0 + 20);
        v16h af, bf0, bf1;
        #pragma unroll
        for (int i = 0; i < 4; ++i) {
            af[i]       = (_Float16)a0[i];
            af[4 + i]   = (_Float16)a1[i];
            af[8 + i]   = (_Float16)a2[i];
            af[12 + i]  = (_Float16)a3[i];
            bf0[i]      = (_Float16)p0[i];
            bf0[4 + i]  = (_Float16)p1[i];
            bf0[8 + i]  = (_Float16)p2[i];
            bf0[12 + i] = (_Float16)p3[i];
            bf1[i]      = (_Float16)q0[i];
            bf1[4 + i]  = (_Float16)q1[i];
            bf1[8 + i]  = (_Float16)q2[i];
            bf1[12 + i] = (_Float16)q3[i];
        }
        acc0 = __builtin_amdgcn_wmma_f32_16x16x32_f16(
            false, af, false, bf0, (short)0, acc0, false, false);
        acc1 = __builtin_amdgcn_wmma_f32_16x16x32_f16(
            false, af, false, bf1, (short)0, acc1, false, false);
    }

    const int col0 = nBase + id16;
    const int col1 = col0 + 16;
    const float bv0 = bias[col0];
    const float bv1 = bias[col1];
    #pragma unroll
    for (int r = 0; r < 8; ++r) {
        int row = mBase + r + hl * 8;          // C/D layout: VGPR r, +8 for upper half-wave
        out[(size_t)row * N + col0] = lrelu(acc0[r] + bv0);
        out[(size_t)row * N + col1] = lrelu(acc1[r] + bv1);
    }
}

// ---------------------------------------------------------------------------
// Sequential scan over t with zero-reset, fused final reorder to (n,c,t,v,m)
// ---------------------------------------------------------------------------
__global__ void scan_out(const float* __restrict__ dcrop, const float* __restrict__ xm,
                         const int* __restrict__ z, float* __restrict__ out)
{
    int idx = blockIdx.x * blockDim.x + threadIdx.x;   // over 256*3*25
    if (idx >= NBATCH * C_IN * V_J) return;
    int v = idx % V_J; int t2 = idx / V_J;
    int c = t2 % C_IN; int nm = t2 / C_IN;
    int n = nm >> 1, m = nm & 1;
    const float* dbase = dcrop + (size_t)(nm * C_IN + c) * T_LEN * V_J + v;
    const float* xbase = xm    + (size_t)(nm * C_IN + c) * T_LEN * V_J + v;
    const int*   zbase = z + nm * T_LEN;
    float* obase = out + (((size_t)(n * C_IN + c) * T_LEN) * V_J + v) * 2 + m;
    float carry = 0.f;
    for (int t = 0; t < T_LEN; ++t) {
        float dt  = (t == 0) ? xbase[0] : dbase[(size_t)t * V_J];
        float fin = zbase[t] ? 0.f : dt + carry;
        carry = fin;
        obase[(size_t)t * (V_J * 2)] = fin;
    }
}

// ---------------------------------------------------------------------------
// Host launcher
// ---------------------------------------------------------------------------
static inline dim3 g1d(long total, int block) { return dim3((unsigned)((total + block - 1) / block)); }

extern "C" void kernel_launch(void* const* d_in, const int* in_sizes, int n_in,
                              void* d_out, int out_size, void* d_ws, size_t ws_size,
                              hipStream_t stream) {
    const float* x     = (const float*)d_in[0];
    const float* dbn_g = (const float*)d_in[1];
    const float* dbn_b = (const float*)d_in[2];
    const float* c1_w  = (const float*)d_in[3];
    const float* c1_b  = (const float*)d_in[4];
    const float* bn1_g = (const float*)d_in[5];
    const float* bn1_b = (const float*)d_in[6];
    const float* c2_w  = (const float*)d_in[7];
    const float* c2_b  = (const float*)d_in[8];
    const float* bn2_g = (const float*)d_in[9];
    const float* bn2_b = (const float*)d_in[10];
    const float* c3_w  = (const float*)d_in[11];
    const float* c3_b  = (const float*)d_in[12];
    const float* bn3_g = (const float*)d_in[13];
    const float* bn3_b = (const float*)d_in[14];
    const float* fc1_w = (const float*)d_in[15];
    const float* fc1_b = (const float*)d_in[16];
    const float* fc2_w = (const float*)d_in[17];
    const float* fc2_b = (const float*)d_in[18];
    const float* fc3_w = (const float*)d_in[19];
    const float* fc3_b = (const float*)d_in[20];
    const float* fc4_w = (const float*)d_in[21];
    const float* fc4_b = (const float*)d_in[22];
    const float* ct1_w = (const float*)d_in[23];
    const float* ct1_b = (const float*)d_in[24];
    const float* bn4_g = (const float*)d_in[25];
    const float* bn4_b = (const float*)d_in[26];
    const float* ct2_w = (const float*)d_in[27];
    const float* ct2_b = (const float*)d_in[28];
    const float* bn5_g = (const float*)d_in[29];
    const float* bn5_b = (const float*)d_in[30];
    const float* ct3_w = (const float*)d_in[31];
    const float* ct3_b = (const float*)d_in[32];
    float* out = (float*)d_out;

    // Workspace layout (floats, regions 16B-aligned; reuse across lifetimes)
    float* ws = (float*)d_ws;
    float* xm   = ws;                          // 5,760,000  (persists)
    int*   zm   = (int*)(ws + 5760000);        //    76,800 ints (persists)
    float* bufA = ws + 5836800;                // 6,225,920  dm -> h3 -> g1
    float* bufB = ws + 12062720;               // 7,987,200  h1 -> f4 -> dcrop
    float* bufC = ws + 20049920;               // 9,961,472  h2 -> g2
    float* bufD = ws + 30011392;               //   262,144  f1 -> f3
    float* bufE = ws + 30273536;               //    32,768  f2

    const int BK = 256;

    // 1) preprocess
    prep_xm<<<g1d((long)NBATCH*C_IN*T_LEN*V_J, BK), BK, 0, stream>>>(x, dbn_g, dbn_b, xm);
    prep_dm<<<g1d((long)NBATCH*C_IN*(T_LEN-1)*V_J, BK), BK, 0, stream>>>(xm, bufA);
    zmask_kernel<<<g1d((long)NBATCH*T_LEN, BK), BK, 0, stream>>>(bufA, zm);

    // 2) encoder convs
    conv_s2_bn_lrelu<<<g1d((long)NBATCH*16*150*13, BK), BK, 0, stream>>>(
        bufA, c1_w, c1_b, bn1_g, bn1_b, bufB, NBATCH, 3, 299, 25, 16, 150, 13);
    conv_s2_bn_lrelu<<<g1d((long)NBATCH*32*75*7, BK), BK, 0, stream>>>(
        bufB, c2_w, c2_b, bn2_g, bn2_b, bufC, NBATCH, 16, 150, 13, 32, 75, 7);
    conv_s2_bn_lrelu<<<g1d((long)NBATCH*64*38*4, BK), BK, 0, stream>>>(
        bufC, c3_w, c3_b, bn3_g, bn3_b, bufA, NBATCH, 32, 75, 7, 64, 38, 4);

    // 3) FC stack via WMMA (M=256 rows; 128 cols/block; grids exact, no divergence)
    fc_wmma_lrelu<<<dim3(1024/128, NBATCH/16), 128, 0, stream>>>(bufA, fc1_w, fc1_b, bufD, 1024, FLATK);
    fc_wmma_lrelu<<<dim3(128/128,  NBATCH/16), 128, 0, stream>>>(bufD, fc2_w, fc2_b, bufE, 128, 1024);
    fc_wmma_lrelu<<<dim3(1024/128, NBATCH/16), 128, 0, stream>>>(bufE, fc3_w, fc3_b, bufD, 1024, 128);
    fc_wmma_lrelu<<<dim3(FLATK/128, NBATCH/16), 128, 0, stream>>>(bufD, fc4_w, fc4_b, bufB, FLATK, 1024);

    // 4) decoder transposed convs
    convT_bn_lrelu<<<g1d((long)NBATCH*32*76*8, BK), BK, 0, stream>>>(
        bufB, ct1_w, ct1_b, bn4_g, bn4_b, bufA, NBATCH, 64, 38, 4, 32, 76, 8);
    convT_bn_lrelu<<<g1d((long)NBATCH*16*152*16, BK), BK, 0, stream>>>(
        bufA, ct2_w, ct2_b, bn5_g, bn5_b, bufC, NBATCH, 32, 76, 8, 16, 152, 16);
    convT3_tanh<<<g1d((long)NBATCH*C_IN*T_LEN*V_J, BK), BK, 0, stream>>>(bufC, ct3_w, ct3_b, bufB);

    // 5) integrate + final layout
    scan_out<<<g1d((long)NBATCH*C_IN*V_J, BK), BK, 0, stream>>>(bufB, xm, zm, out);
}
